// TripletBlock_15848429322412
// MI455X (gfx1250) — matrix operand order repaired and checked by reading, not compiled, not run on hardware
//
#include <hip/hip_runtime.h>

// ---------------------------------------------------------------------------
// CDNA5 / gfx1250 TripletBlock GNN.
// Memory-bound (~2 GB traffic vs 132 GFLOP): gather f16 mirrors of the
// activations straight into LDS with global_load_async_to_lds_b128
// (ASYNCcnt), matmul with v_wmma_f32_16x16x32_f16 (f32 accumulate).
// ---------------------------------------------------------------------------

typedef __attribute__((ext_vector_type(16))) _Float16 v16h;
typedef __attribute__((ext_vector_type(8)))  _Float16 v8h;
typedef __attribute__((ext_vector_type(8)))  float    v8f;

#define NL   2
#define NN_  50000
#define NE_  300000
#define NA_  600000
#define ND_  128
#define ED_  64
#define AD_  64
#define NG_  16
#define EPSN 1e-5f

// ------------------------------ utility kernels ----------------------------

__global__ void fill_f32(float* __restrict__ p, long n) {
    long i = (long)blockIdx.x * blockDim.x + threadIdx.x;
    long s = (long)gridDim.x * blockDim.x;
    for (; i < n; i += s) p[i] = 0.0f;
}

__global__ void fill_i32(int* __restrict__ p, int n) {
    int i = blockIdx.x * blockDim.x + threadIdx.x;
    if (i < n) p[i] = 0;
}

// copy f32 source into an f32 working buffer + f16 mirror
__global__ void copy_cvt(float* __restrict__ d32, _Float16* __restrict__ d16,
                         const float* __restrict__ s, long n) {
    long i = (long)blockIdx.x * blockDim.x + threadIdx.x;
    long st = (long)gridDim.x * blockDim.x;
    for (; i < n; i += st) {
        float v = s[i];
        d32[i] = v;
        d16[i] = (_Float16)v;
    }
}

__global__ void count_batch(const int* __restrict__ batch, int* __restrict__ cnt, int rows) {
    int i = blockIdx.x * blockDim.x + threadIdx.x;
    int s = gridDim.x * blockDim.x;
    for (; i < rows; i += s) atomicAdd(&cnt[batch[i]], 1);
}

// Pack row-major f32 W[K][Nn] into f16 B-fragment order:
// [kt][nt][lane][16 halves], k = kt*32 + (lane>=16)*8 + (h&7) + (h>=8)*16, n = nt*16 + (lane&15)
__global__ void pack_w(const float* __restrict__ W, _Float16* __restrict__ Wp,
                       int K, int Nn) {
    const int nT = Nn >> 4;
    long total = (long)K * Nn;
    long i = (long)blockIdx.x * blockDim.x + threadIdx.x;
    long s = (long)gridDim.x * blockDim.x;
    for (; i < total; i += s) {
        long tile = i >> 9;
        int  within = (int)(i & 511);
        int  lane = within >> 4;
        int  h    = within & 15;
        int  nt   = (int)(tile % nT);
        int  kt   = (int)(tile / nT);
        int  k = kt * 32 + ((lane >> 4) << 3) + (h & 7) + ((h >> 3) << 4);
        int  n = (nt << 4) + (lane & 15);
        Wp[i] = (_Float16)W[(long)k * Nn + n];
    }
}

// ------------------------------ gather-GEMM --------------------------------
// A tile (16 rows) = concat(S0[idx0[m]], S1[idx1[m]], S2[m]) in f16,
// staged into LDS via async global->LDS B128 transfers.
// tgt == nullptr : residual epilogue, out[m] += acc + bias (out is f32)
// tgt != nullptr : msg = relu(acc + bias); atomicAdd(out[tgt[m]], msg)

__global__ __launch_bounds__(128) void gather_gemm(
    const _Float16* __restrict__ S0, const int* __restrict__ idx0, int f0,
    const _Float16* __restrict__ S1, const int* __restrict__ idx1, int f1,
    const _Float16* __restrict__ S2, int f2,
    const _Float16* __restrict__ Wp, const float* __restrict__ bias,
    float* __restrict__ out, const int* __restrict__ tgt,
    int M, int K, int NDIM)
{
    extern __shared__ _Float16 ldsA[];   // 16 rows x (K+8) halves, padded
    const int strideH = K + 8;
    const int m0  = blockIdx.x << 4;
    const int tid = threadIdx.x;
    const int f01 = f0 + f1;

    // async-gather the f16 A tile into LDS, 16B per lane per transfer
    const int chunksPerRow = K >> 3;            // 8 halves = 16B per chunk
    const int totalChunks  = chunksPerRow << 4; // multiple of 128 here
    for (int c = tid; c < totalChunks; c += 128) {
        const int r  = c / chunksPerRow;
        const int ch = (c - r * chunksPerRow) << 3;   // half offset in row
        int m = m0 + r; if (m >= M) m = M - 1;        // defensive clamp
        const _Float16* gsrc;
        if (ch < f0)       gsrc = S0 + (long)idx0[m] * f0 + ch;
        else if (ch < f01) gsrc = S1 + (long)idx1[m] * f1 + (ch - f0);
        else               gsrc = S2 + (long)m * f2 + (ch - f01);
        const unsigned int loff = (unsigned int)(uintptr_t)(ldsA + r * strideH + ch);
        asm volatile("global_load_async_to_lds_b128 %0, %1, off"
                     :: "v"(loff), "v"((unsigned long long)(uintptr_t)gsrc)
                     : "memory");
    }
    asm volatile("s_wait_asynccnt 0x0" ::: "memory");
    __syncthreads();

    const int wave = tid >> 5;
    const int lane = tid & 31;
    const int nTiles = NDIM >> 4;
    const int kTiles = K >> 5;
    const int arow = lane & 15;
    const int kb   = (lane >> 4) << 3;          // A frag: K base 0 or 8
    const _Float16* lrow = ldsA + arow * strideH + kb;

    for (int nt = wave; nt < nTiles; nt += 4) {
        v8f acc = {};
        const _Float16* wpp = Wp + ((long)nt * 32 + lane) * 16;
        const long kstride = (long)nTiles * 512;
        for (int kt = 0; kt < kTiles; ++kt) {
            union { v16h v; v8h h[2]; } af, bf;
            af.h[0] = *(const v8h*)(lrow + kt * 32);        // K[kb .. kb+7]
            af.h[1] = *(const v8h*)(lrow + kt * 32 + 16);   // K[kb+16 .. kb+23]
            bf.v    = *(const v16h*)(wpp + kt * kstride);   // packed B fragment
            acc = __builtin_amdgcn_wmma_f32_16x16x32_f16(
                false, af.v, false, bf.v, (short)0, acc, false, false);
        }
        const int col = (nt << 4) + (lane & 15);
        const float bv = bias[col];
        const int mb = m0 + ((lane >> 4) << 3);   // D: lanes>=16 hold M = j+8
        if (tgt) {
#pragma unroll
            for (int j = 0; j < 8; ++j) {
                const int m = mb + j;
                if (m < M) {
                    float v = acc[j] + bv;
                    v = v > 0.0f ? v : 0.0f;
                    atomicAdd(&out[(long)tgt[m] * NDIM + col], v);
                }
            }
        } else {
#pragma unroll
            for (int j = 0; j < 8; ++j) {
                const int m = mb + j;
                if (m < M) {
                    const long o = (long)m * NDIM + col;
                    out[o] = out[o] + acc[j] + bv;
                }
            }
        }
    }
}

// --------------------------- graph-norm kernels ----------------------------

__global__ void stats_k(float* __restrict__ h, const float* __restrict__ addin,
                        const int* __restrict__ batch, float* __restrict__ stats,
                        int rows, int fsh)
{
    __shared__ float ss[NG_], sq[NG_];
    if (threadIdx.x < NG_) { ss[threadIdx.x] = 0.0f; sq[threadIdx.x] = 0.0f; }
    __syncthreads();
    const long total = (long)rows << fsh;
    long i = (long)blockIdx.x * blockDim.x + threadIdx.x;
    const long s = (long)gridDim.x * blockDim.x;
    for (; i < total; i += s) {
        float v = h[i];
        if (addin) { v += addin[i]; h[i] = v; }
        const int g = batch[i >> fsh];
        atomicAdd(&ss[g], v);
        atomicAdd(&sq[g], v * v);
    }
    __syncthreads();
    if (threadIdx.x < NG_) {
        atomicAdd(&stats[threadIdx.x],        ss[threadIdx.x]);
        atomicAdd(&stats[NG_ + threadIdx.x],  sq[threadIdx.x]);
    }
}

// normalize in place and refresh the f16 mirror
__global__ void apply_k(float* __restrict__ h, _Float16* __restrict__ mirror,
                        const int* __restrict__ batch,
                        const float* __restrict__ stats, const int* __restrict__ cnt,
                        const float* __restrict__ w, const float* __restrict__ b,
                        int rows, int F, int fsh)
{
    const long total = (long)rows << fsh;
    long i = (long)blockIdx.x * blockDim.x + threadIdx.x;
    const long s = (long)gridDim.x * blockDim.x;
    for (; i < total; i += s) {
        const int r = (int)(i >> fsh);
        const int c = (int)(i & (F - 1));
        const int g = batch[r];
        const float norm = fmaxf((float)cnt[g], 1.0f) * (float)F;
        const float mean = stats[g] / norm;
        float var = stats[NG_ + g] / norm - mean * mean;
        var = var > 0.0f ? var : 0.0f;
        const float inv = rsqrtf(var + EPSN);
        const float res = (h[i] - mean) * inv * w[c] + b[c];
        h[i] = res;
        mirror[i] = (_Float16)res;
    }
}

// ------------------------------- host side ---------------------------------

extern "C" void kernel_launch(void* const* d_in, const int* in_sizes, int n_in,
                              void* d_out, int out_size, void* d_ws, size_t ws_size,
                              hipStream_t stream) {
    (void)in_sizes; (void)n_in; (void)out_size; (void)ws_size;

    const float* x_in  = (const float*)d_in[0];
    const float* e_in  = (const float*)d_in[1];
    const float* a_in  = (const float*)d_in[2];
    const float* W_ne  = (const float*)d_in[3];
    const float* b_ne  = (const float*)d_in[4];
    const float* g_e   = (const float*)d_in[5];
    const float* be_e  = (const float*)d_in[6];
    const float* W_ea  = (const float*)d_in[7];
    const float* b_ea  = (const float*)d_in[8];
    const float* g_a   = (const float*)d_in[9];
    const float* be_a  = (const float*)d_in[10];
    const float* W_emp = (const float*)d_in[11];
    const float* b_emp = (const float*)d_in[12];
    const float* g_emp = (const float*)d_in[13];
    const float* be_emp= (const float*)d_in[14];
    const float* W_nmp = (const float*)d_in[15];
    const float* b_nmp = (const float*)d_in[16];
    const float* g_nmp = (const float*)d_in[17];
    const float* be_nmp= (const float*)d_in[18];
    const int* node_batch = (const int*)d_in[19];
    const int* eidx       = (const int*)d_in[20];
    const int* edge_batch = (const int*)d_in[21];
    const int* tbidx      = (const int*)d_in[22];
    const int* angle_batch= (const int*)d_in[23];

    const int* src_e = eidx;
    const int* dst_e = eidx + NE_;
    const int* src_t = tbidx;
    const int* dst_t = tbidx + NA_;

    float* x = (float*)d_out;                       // x evolves in d_out

    // workspace layout
    char* ws = (char*)d_ws;
    float* e_buf = (float*)ws;  ws += (size_t)NE_ * ED_ * 4;
    float* a_buf = (float*)ws;  ws += (size_t)NA_ * AD_ * 4;
    float* agg_e = (float*)ws;  ws += (size_t)NE_ * ED_ * 4;
    float* agg_n = (float*)ws;  ws += (size_t)NN_ * ND_ * 4;
    _Float16* x16 = (_Float16*)ws;  ws += (size_t)NN_ * ND_ * 2;
    _Float16* e16 = (_Float16*)ws;  ws += (size_t)NE_ * ED_ * 2;
    _Float16* a16 = (_Float16*)ws;  ws += (size_t)NA_ * AD_ * 2;
    _Float16* wp = (_Float16*)ws;
    const long WP_NE = 320 * 64, WP_EA = 192 * 64, WP_EMP = 192 * 64, WP_NMP = 320 * 128;
    _Float16* wpNE  = wp;
    _Float16* wpEA  = wpNE  + NL * WP_NE;
    _Float16* wpEMP = wpEA  + NL * WP_EA;
    _Float16* wpNMP = wpEMP + NL * WP_EMP;
    ws += (size_t)(NL * (WP_NE + WP_EA + WP_EMP + WP_NMP)) * 2;
    ws = (char*)(((size_t)ws + 255) & ~(size_t)255);
    int*   cnt   = (int*)ws;    ws += 3 * NG_ * 4;
    float* stats = (float*)ws;  // 2*NG_ floats

    int* cnt_node  = cnt;
    int* cnt_edge  = cnt + NG_;
    int* cnt_angle = cnt + 2 * NG_;

    const dim3 EB(256);
    const dim3 EG(2048);

    // init working buffers + f16 mirrors from (const) inputs
    copy_cvt<<<EG, EB, 0, stream>>>(x,     x16, x_in, (long)NN_ * ND_);
    copy_cvt<<<EG, EB, 0, stream>>>(e_buf, e16, e_in, (long)NE_ * ED_);
    copy_cvt<<<EG, EB, 0, stream>>>(a_buf, a16, a_in, (long)NA_ * AD_);

    // per-graph row counts
    fill_i32<<<1, 64, 0, stream>>>(cnt, 3 * NG_);
    count_batch<<<256, 256, 0, stream>>>(node_batch,  cnt_node,  NN_);
    count_batch<<<256, 256, 0, stream>>>(edge_batch,  cnt_edge,  NE_);
    count_batch<<<512, 256, 0, stream>>>(angle_batch, cnt_angle, NA_);

    // pack all weights to f16 B-fragment layout
    for (int i = 0; i < NL; ++i) {
        pack_w<<<128, 256, 0, stream>>>(W_ne  + (long)i * WP_NE,  wpNE  + (long)i * WP_NE,  320, 64);
        pack_w<<<128, 256, 0, stream>>>(W_ea  + (long)i * WP_EA,  wpEA  + (long)i * WP_EA,  192, 64);
        pack_w<<<128, 256, 0, stream>>>(W_emp + (long)i * WP_EMP, wpEMP + (long)i * WP_EMP, 192, 64);
        pack_w<<<256, 256, 0, stream>>>(W_nmp + (long)i * WP_NMP, wpNMP + (long)i * WP_NMP, 320, 128);
    }

    const int gridE = (NE_ + 15) / 16;
    const int gridA = (NA_ + 15) / 16;
    const size_t sh320 = (size_t)16 * (320 + 8) * 2;
    const size_t sh192 = (size_t)16 * (192 + 8) * 2;

    for (int i = 0; i < NL; ++i) {
        // --- stage 1: node-pair -> edge update (+residual, graph norm) ---
        fill_f32<<<1, 64, 0, stream>>>(stats, 2 * NG_);
        gather_gemm<<<gridE, 128, sh320, stream>>>(
            x16, src_e, ND_, x16, dst_e, ND_, e16, ED_,
            wpNE + (long)i * WP_NE, b_ne + (long)i * ED_,
            e_buf, nullptr, NE_, 320, ED_);
        stats_k<<<EG, EB, 0, stream>>>(e_buf, nullptr, edge_batch, stats, NE_, 6);
        apply_k<<<EG, EB, 0, stream>>>(e_buf, e16, edge_batch, stats, cnt_edge,
                                       g_e + (long)i * ED_, be_e + (long)i * ED_, NE_, ED_, 6);

        // --- stage 2: edge-pair -> angle update (+residual, graph norm) ---
        fill_f32<<<1, 64, 0, stream>>>(stats, 2 * NG_);
        gather_gemm<<<gridA, 128, sh192, stream>>>(
            e16, src_t, ED_, e16, dst_t, ED_, a16, AD_,
            wpEA + (long)i * WP_EA, b_ea + (long)i * AD_,
            a_buf, nullptr, NA_, 192, AD_);
        stats_k<<<EG, EB, 0, stream>>>(a_buf, nullptr, angle_batch, stats, NA_, 6);
        apply_k<<<EG, EB, 0, stream>>>(a_buf, a16, angle_batch, stats, cnt_angle,
                                       g_a + (long)i * AD_, be_a + (long)i * AD_, NA_, AD_, 6);

        // --- stage 3: angle-mediated message passing over edges ---
        fill_f32<<<EG, EB, 0, stream>>>(agg_e, (long)NE_ * ED_);
        fill_f32<<<1, 64, 0, stream>>>(stats, 2 * NG_);
        gather_gemm<<<gridA, 128, sh192, stream>>>(
            e16, src_t, ED_, e16, dst_t, ED_, a16, AD_,
            wpEMP + (long)i * WP_EMP, b_emp + (long)i * ED_,
            agg_e, dst_t, NA_, 192, ED_);
        stats_k<<<EG, EB, 0, stream>>>(e_buf, agg_e, edge_batch, stats, NE_, 6);
        apply_k<<<EG, EB, 0, stream>>>(e_buf, e16, edge_batch, stats, cnt_edge,
                                       g_emp + (long)i * ED_, be_emp + (long)i * ED_, NE_, ED_, 6);

        // --- stage 4: edge-mediated message passing over nodes ---
        fill_f32<<<EG, EB, 0, stream>>>(agg_n, (long)NN_ * ND_);
        fill_f32<<<1, 64, 0, stream>>>(stats, 2 * NG_);
        gather_gemm<<<gridE, 128, sh320, stream>>>(
            x16, src_e, ND_, x16, dst_e, ND_, e16, ED_,
            wpNMP + (long)i * WP_NMP, b_nmp + (long)i * ND_,
            agg_n, dst_e, NE_, 320, ND_);
        stats_k<<<EG, EB, 0, stream>>>(x, agg_n, node_batch, stats, NN_, 7);
        apply_k<<<EG, EB, 0, stream>>>(x, x16, node_batch, stats, cnt_node,
                                       g_nmp + (long)i * ND_, be_nmp + (long)i * ND_, NN_, ND_, 7);
    }
}
